// ParametrizationNet_26182120636661
// MI455X (gfx1250) — matrix operand order repaired and checked by reading, not compiled, not run on hardware
//
#include <hip/hip_runtime.h>

// Neural-ODE RK4 integrator for MI455X (gfx1250, wave32, WMMA).
// Each workgroup (512 threads = 16 waves) owns 64 batch rows for the WHOLE
// 32-step integration. All activations live in LDS (312KB of the 320KB/WGP);
// only f16-packed weights (L2-resident, ~6.3MB) stream in per layer.
// GEMMs use v_wmma_f32_16x16x32_f16 with f32 accumulation; bias + t*time_row
// are folded into the accumulator init.
// v2: split each wave's 64-col slab into two N-passes (acc[4][2] = 64 VGPRs
// live instead of 128) to eliminate the scratch spills seen in v1's inner loop.

typedef _Float16 half_t;
typedef __attribute__((ext_vector_type(16))) _Float16 v16h;
typedef __attribute__((ext_vector_type(8)))  _Float16 v8h;   // 16B = one b128
typedef __attribute__((ext_vector_type(8)))  float    v8f;

#define WAVES   16
#define BLOCK   (WAVES * 32)
#define MROWS   64        // batch rows per workgroup
#define HID     1024
#define VDIM    64        // var_dim 50 padded to 64
#define NSTEPS  32

__device__ __forceinline__ v8f splat8(float x) {
  v8f v;
#pragma unroll
  for (int i = 0; i < 8; ++i) v[i] = x;
  return v;
}

// A-fragment (16x32 f16) per ISA layout:
// lane<16: M=lane,  halves 0-7 = K(ks*32+0..7),  halves 8-15 = K(+16..23)
// lane>=16: M=lane-16, base K offset +8.
__device__ __forceinline__ v16h load_a_frag(const half_t* __restrict__ act,
                                            int row_stride_h, int row,
                                            int ks, int lane) {
  int base = row * row_stride_h + ks * 32 + ((lane >> 4) << 3);
  v16h a;
  *reinterpret_cast<v8h*>(&a)       = *reinterpret_cast<const v8h*>(act + base);
  *(reinterpret_cast<v8h*>(&a) + 1) = *reinterpret_cast<const v8h*>(act + base + 16);
  return a;
}

// ---- hidden layer: [64 x 1024] = relu([64 x 1024] @ [1024 x 1024] + b + t*trow)
// Two N-passes of 32 columns each to keep live accumulators at 64 VGPRs.
__device__ __forceinline__ void layer_hidden(const half_t* __restrict__ actIn,
                                             half_t* __restrict__ actOut,
                                             const half_t* __restrict__ wf,
                                             const float* __restrict__ bias,
                                             const float* __restrict__ trow,
                                             float t, int wave, int lane) {
  const int n_lane = lane & 15;
  const int hi     = lane >> 4;
#pragma unroll 1
  for (int pass = 0; pass < 2; ++pass) {
    const int nt0 = wave * 4 + pass * 2;       // first of 2 n-tiles this pass
    v8f acc[4][2];
#pragma unroll
    for (int j = 0; j < 2; ++j) {
      int n = (nt0 + j) * 16 + n_lane;
      float c0 = bias[n] + t * trow[n];
#pragma unroll
      for (int m = 0; m < 4; ++m) acc[m][j] = splat8(c0);
    }
#pragma unroll 1
    for (int ks = 0; ks < 32; ++ks) {
      v16h a[4];
#pragma unroll
      for (int m = 0; m < 4; ++m)
        a[m] = load_a_frag(actIn, HID, m * 16 + n_lane, ks, lane);
#pragma unroll
      for (int j = 0; j < 2; ++j) {
        v16h b = *reinterpret_cast<const v16h*>(
            wf + ((size_t)((nt0 + j) * 32 + ks) * 32 + lane) * 16);
#pragma unroll
        for (int m = 0; m < 4; ++m)
          acc[m][j] = __builtin_amdgcn_wmma_f32_16x16x32_f16(
              false, a[m], false, b, (short)0, acc[m][j], false, false);
      }
    }
#pragma unroll
    for (int m = 0; m < 4; ++m)
#pragma unroll
      for (int j = 0; j < 2; ++j) {
        int col  = (nt0 + j) * 16 + n_lane;
        int row0 = m * 16 + hi * 8;
#pragma unroll
        for (int r = 0; r < 8; ++r) {
          float v = acc[m][j][r];
          v = v > 0.f ? v : 0.f;                 // ReLU
          actOut[(row0 + r) * HID + col] = (half_t)v;
        }
      }
  }
}

// ---- input layer: [64 x 1024] = relu([64 x 64pad] @ [64pad x 1024] + b + t*trow)
__device__ __forceinline__ void layer_in(const half_t* __restrict__ ytmp16,
                                         half_t* __restrict__ actOut,
                                         const half_t* __restrict__ wf,
                                         const float* __restrict__ bias,
                                         const float* __restrict__ trow,
                                         float t, int wave, int lane) {
  const int n_lane = lane & 15;
  const int hi     = lane >> 4;
#pragma unroll 1
  for (int pass = 0; pass < 2; ++pass) {
    const int nt0 = wave * 4 + pass * 2;
    v8f acc[4][2];
#pragma unroll
    for (int j = 0; j < 2; ++j) {
      int n = (nt0 + j) * 16 + n_lane;
      float c0 = bias[n] + t * trow[n];
#pragma unroll
      for (int m = 0; m < 4; ++m) acc[m][j] = splat8(c0);
    }
#pragma unroll
    for (int ks = 0; ks < 2; ++ks) {
      v16h a[4];
#pragma unroll
      for (int m = 0; m < 4; ++m)
        a[m] = load_a_frag(ytmp16, VDIM, m * 16 + n_lane, ks, lane);
#pragma unroll
      for (int j = 0; j < 2; ++j) {
        v16h b = *reinterpret_cast<const v16h*>(
            wf + ((size_t)((nt0 + j) * 2 + ks) * 32 + lane) * 16);
#pragma unroll
        for (int m = 0; m < 4; ++m)
          acc[m][j] = __builtin_amdgcn_wmma_f32_16x16x32_f16(
              false, a[m], false, b, (short)0, acc[m][j], false, false);
      }
    }
#pragma unroll
    for (int m = 0; m < 4; ++m)
#pragma unroll
      for (int j = 0; j < 2; ++j) {
        int col  = (nt0 + j) * 16 + n_lane;
        int row0 = m * 16 + hi * 8;
#pragma unroll
        for (int r = 0; r < 8; ++r) {
          float v = acc[m][j][r];
          v = v > 0.f ? v : 0.f;
          actOut[(row0 + r) * HID + col] = (half_t)v;
        }
      }
  }
}

// ---- output layer: kcur[64 x 64pad] = [64 x 1024] @ [1024 x 64pad] + b + t*trow (no relu)
__device__ __forceinline__ void layer_out(const half_t* __restrict__ actIn,
                                          float* __restrict__ kcur,
                                          const half_t* __restrict__ wf,
                                          const float* __restrict__ bias,
                                          const float* __restrict__ trow,
                                          float t, int wave, int lane) {
  const int n_lane = lane & 15;
  const int hi     = lane >> 4;
  const int mt     = wave >> 2;    // 4 M-tiles
  const int nt     = wave & 3;     // 4 N-tiles (64 padded cols)
  int n = nt * 16 + n_lane;
  float c0 = (n < 50) ? (bias[n] + t * trow[n]) : 0.f;
  v8f acc = splat8(c0);
#pragma unroll 1
  for (int ks = 0; ks < 32; ++ks) {
    v16h a = load_a_frag(actIn, HID, mt * 16 + n_lane, ks, lane);
    v16h b = *reinterpret_cast<const v16h*>(
        wf + ((size_t)(nt * 32 + ks) * 32 + lane) * 16);
    acc = __builtin_amdgcn_wmma_f32_16x16x32_f16(
        false, a, false, b, (short)0, acc, false, false);
  }
  int row0 = mt * 16 + hi * 8;
#pragma unroll
  for (int r = 0; r < 8; ++r)
    kcur[(row0 + r) * VDIM + n] = acc[r];
}

extern "C" __global__ void __launch_bounds__(BLOCK, 1)
ode_rk4_kernel(const float* __restrict__ x,
               const float* __restrict__ b0, const float* __restrict__ tr0,
               const float* __restrict__ b1, const float* __restrict__ tr1,
               const float* __restrict__ b2, const float* __restrict__ tr2,
               const float* __restrict__ b3, const float* __restrict__ tr3,
               const float* __restrict__ b4, const float* __restrict__ tr4,
               const half_t* __restrict__ wf0, const half_t* __restrict__ wf1,
               const half_t* __restrict__ wf2, const half_t* __restrict__ wf3,
               const half_t* __restrict__ wf4,
               float* __restrict__ out) {
  extern __shared__ char smem[];
  half_t* actA  = (half_t*)smem;                 // 64*1024 f16 = 128KB
  half_t* actB  = actA + MROWS * HID;            // 128KB
  half_t* ytmp  = actB + MROWS * HID;            // 64*64 f16 = 8KB
  float*  yst   = (float*)(ytmp + MROWS * VDIM); // 16KB  (state y)
  float*  racc  = yst  + MROWS * VDIM;           // 16KB  (RK4 accumulator)
  float*  kcur  = racc + MROWS * VDIM;           // 16KB  (current k)

  const int tid   = threadIdx.x;
  const int wave  = tid >> 5;
  const int lane  = tid & 31;
  const int rowg0 = blockIdx.x * MROWS;
  const int e0    = tid * 8;                     // 512 threads * 8 = 4096 elems

  // ---- init state: y[:, :2] = x, rest 0 (var_dim padded 50->64)
#pragma unroll
  for (int j = 0; j < 8; ++j) {
    int e = e0 + j, row = e >> 6, col = e & 63;
    float v = (col < 2) ? x[(size_t)(rowg0 + row) * 2 + col] : 0.f;
    yst[e]  = v;
    ytmp[e] = (half_t)v;
  }
  __syncthreads();

  const float dt = 1.0f / NSTEPS;

#pragma unroll 1
  for (int step = 0; step < NSTEPS; ++step) {
    float tb = step * dt;
#pragma unroll 1
    for (int s = 0; s < 4; ++s) {
      float ts = tb + ((s == 0) ? 0.f : (s == 3) ? dt : 0.5f * dt);

      // k_s = odefunc(ts, ytmp)
      layer_in    (ytmp, actA, wf0, b0, tr0, ts, wave, lane); __syncthreads();
      layer_hidden(actA, actB, wf1, b1, tr1, ts, wave, lane); __syncthreads();
      layer_hidden(actB, actA, wf2, b2, tr2, ts, wave, lane); __syncthreads();
      layer_hidden(actA, actB, wf3, b3, tr3, ts, wave, lane); __syncthreads();
      layer_out   (actB, kcur, wf4, b4, tr4, ts, wave, lane); __syncthreads();

      // RK4 stage combine (f32 in LDS)
#pragma unroll
      for (int j = 0; j < 8; ++j) {
        int e = e0 + j;
        float k  = kcur[e];
        float yv = yst[e];
        if (s == 0) {
          racc[e] = yv + (dt / 6.f) * k;
          ytmp[e] = (half_t)(yv + 0.5f * dt * k);
        } else if (s == 1) {
          racc[e] += (dt / 3.f) * k;
          ytmp[e]  = (half_t)(yv + 0.5f * dt * k);
        } else if (s == 2) {
          racc[e] += (dt / 3.f) * k;
          ytmp[e]  = (half_t)(yv + dt * k);
        } else {
          float yn = racc[e] + (dt / 6.f) * k;
          yst[e]  = yn;
          ytmp[e] = (half_t)yn;
        }
      }
      __syncthreads();
    }
  }

  // ---- emit y1[:, :3]
#pragma unroll
  for (int j = 0; j < 8; ++j) {
    int e = e0 + j, row = e >> 6, col = e & 63;
    if (col < 3) out[(size_t)(rowg0 + row) * 3 + col] = yst[e];
  }
}

// Pack f32 weights -> f16 WMMA B-fragments:
// frag[nt][ks][lane][c] = W[ks*32 + 8*(lane>>4) + (c&7) + 16*(c>>3)][nt*16 + (lane&15)]   (zero-padded)
// so the main loop fetches one contiguous 32B chunk per lane.
extern "C" __global__ void pack_weights(const float* __restrict__ W,
                                        half_t* __restrict__ dst,
                                        int Kreal, int Kpad, int Nreal,
                                        int Ntiles) {
  int idx = blockIdx.x * blockDim.x + threadIdx.x;
  if (idx >= Kpad * Ntiles) return;
  int k  = idx % Kpad;
  int nt = idx / Kpad;
  int ks = k >> 5, lane = k & 31;
  half_t* d = dst + ((size_t)(nt * (Kpad >> 5) + ks) * 32 + lane) * 16;
#pragma unroll
  for (int c = 0; c < 16; ++c) {
    int n = nt * 16 + (lane & 15);
    int kk = ks * 32 + ((lane >> 4) << 3) + (c & 7) + ((c >> 3) << 4);
    float v = (kk < Kreal && n < Nreal) ? W[(size_t)kk * Nreal + n] : 0.f;
    d[c] = (half_t)v;
  }
}

extern "C" void kernel_launch(void* const* d_in, const int* in_sizes, int n_in,
                              void* d_out, int out_size, void* d_ws, size_t ws_size,
                              hipStream_t stream) {
  (void)in_sizes; (void)n_in; (void)out_size; (void)ws_size;
  const float* x  = (const float*)d_in[0];
  const float* W0 = (const float*)d_in[1]; const float* b0 = (const float*)d_in[2];
  const float* W1 = (const float*)d_in[3]; const float* b1 = (const float*)d_in[4];
  const float* W2 = (const float*)d_in[5]; const float* b2 = (const float*)d_in[6];
  const float* W3 = (const float*)d_in[7]; const float* b3 = (const float*)d_in[8];
  const float* W4 = (const float*)d_in[9]; const float* b4 = (const float*)d_in[10];

  // f16 packed-weight workspace layout (~6.5MB, L2-resident)
  half_t* wf0 = (half_t*)d_ws;                      // 64nt * 2ks  *32*16 = 65536
  half_t* wf1 = wf0 + (size_t)64 * 2  * 32 * 16;
  half_t* wf2 = wf1 + (size_t)64 * 32 * 32 * 16;    // 1048576 each
  half_t* wf3 = wf2 + (size_t)64 * 32 * 32 * 16;
  half_t* wf4 = wf3 + (size_t)64 * 32 * 32 * 16;    // 4nt * 32ks *32*16 = 65536

  // Prologue: repack weights (deterministic, every launch)
  pack_weights<<<(64 * 64    + 255) / 256, 256, 0, stream>>>(W0, wf0,   50,   64, 1024, 64);
  pack_weights<<<(1024 * 64  + 255) / 256, 256, 0, stream>>>(W1, wf1, 1024, 1024, 1024, 64);
  pack_weights<<<(1024 * 64  + 255) / 256, 256, 0, stream>>>(W2, wf2, 1024, 1024, 1024, 64);
  pack_weights<<<(1024 * 64  + 255) / 256, 256, 0, stream>>>(W3, wf3, 1024, 1024, 1024, 64);
  pack_weights<<<(1024 * 4   + 255) / 256, 256, 0, stream>>>(W4, wf4, 1024, 1024,   50,  4);

  // time rows are the last row of each (d1+1, d2) weight matrix (contiguous)
  const float* tr0 = W0 + (size_t)50   * 1024;
  const float* tr1 = W1 + (size_t)1024 * 1024;
  const float* tr2 = W2 + (size_t)1024 * 1024;
  const float* tr3 = W3 + (size_t)1024 * 1024;
  const float* tr4 = W4 + (size_t)1024 * 50;

  // 2*128KB act ping-pong + 8KB f16 state + 3*16KB f32 RK4 state = 319488B
  size_t shmem = (size_t)2 * MROWS * HID * sizeof(half_t)
               + (size_t)MROWS * VDIM * sizeof(half_t)
               + (size_t)3 * MROWS * VDIM * sizeof(float);

  ode_rk4_kernel<<<32768 / MROWS, BLOCK, shmem, stream>>>(
      x, b0, tr0, b1, tr1, b2, tr2, b3, tr3, b4, tr4,
      wf0, wf1, wf2, wf3, wf4, (float*)d_out);
}
